// MultiHeadSelfAttention_44513041056470
// MI455X (gfx1250) — compile-verified, matrix operand order checked by reading, not run on hardware
//
#include <hip/hip_runtime.h>
#include <hip/hip_bf16.h>

typedef __attribute__((ext_vector_type(16))) _Float16 v16h;
typedef __attribute__((ext_vector_type(8)))  _Float16 v8h;
typedef __attribute__((ext_vector_type(8)))  float    v8f;
typedef int b128_t __attribute__((vector_size(16)));   // matches builtin param

#define D_MODEL 1024
#define S_LEN   2048
#define NHEAD   16
#define HDIM    64
#define NROWS   4096   // B * S

// ---------------------------------------------------------------------------
// CDNA5 async global->LDS copies (ASYNCcnt path), with safe fallback
// ---------------------------------------------------------------------------
#if defined(__has_builtin)
#if __has_builtin(__builtin_amdgcn_global_load_async_to_lds_b128) && \
    __has_builtin(__builtin_amdgcn_s_wait_asynccnt)
#define HAVE_ASYNC_LDS 1
#endif
#endif
#ifndef HAVE_ASYNC_LDS
#define HAVE_ASYNC_LDS 0
#endif

// Copy 16 bytes global -> LDS for this lane.
__device__ __forceinline__ void cp16(const _Float16* g, _Float16* l) {
#if HAVE_ASYNC_LDS
  __builtin_amdgcn_global_load_async_to_lds_b128(
      (__attribute__((address_space(1))) b128_t*)(unsigned long long)g,
      (__attribute__((address_space(3))) b128_t*)(unsigned)(unsigned long long)l,
      0, 0);
#else
  *(v8h*)l = *(const v8h*)g;
#endif
}

__device__ __forceinline__ void wait_async_copies() {
#if HAVE_ASYNC_LDS
  __builtin_amdgcn_s_wait_asynccnt(0);
#endif
}

// ---------------------------------------------------------------------------
// WMMA helpers
// ---------------------------------------------------------------------------
__device__ __forceinline__ v8f wmma16(v16h a, v16h b, v8f c) {
  // D = A(16x32 f16) * B(32x16 f16) + C(16x16 f32)
  return __builtin_amdgcn_wmma_f32_16x16x32_f16(
      false, a, false, b, (short)0, c, false, false);
}

// Load a 16x32 f16 A-fragment (or B-fragment via B^T) from a row-major
// source. `p` points at (row_base + khalf*8); per ISA 7.12.2 the lane's
// elements 0-7 sit at K-offset 8*kh and 8-15 at 16+8*kh. 16B alignment req.
__device__ __forceinline__ v16h ldfrag(const _Float16* p) {
  v8h lo = *(const v8h*)p;
  v8h hi = *(const v8h*)(p + 16);
  return __builtin_shufflevector(lo, hi, 0,1,2,3,4,5,6,7,8,9,10,11,12,13,14,15);
}

// ---------------------------------------------------------------------------
// f32 -> f16 conversion kernels
// ---------------------------------------------------------------------------
__global__ void cvt_f32_to_f16(const float* __restrict__ in,
                               _Float16* __restrict__ out, int n) {
  int i = blockIdx.x * blockDim.x + threadIdx.x;
  if (i < n) out[i] = (_Float16)in[i];
}

// W[in][out] (f32) -> Wt[out][in] (f16)
__global__ void cvt_transpose(const float* __restrict__ W,
                              _Float16* __restrict__ Wt, int D) {
  int i = blockIdx.x * blockDim.x + threadIdx.x;
  if (i < D * D) {
    int o = i / D, in_ = i - o * D;
    Wt[i] = (_Float16)W[(size_t)in_ * D + o];
  }
}

// ---------------------------------------------------------------------------
// WMMA GEMM: C[M,N] = A[M,K](f16,row-major) * Bt[N,K]^T + bias[N]
//   mode 0: store f16 row-major [M,N]
//   mode 1: store f16 transposed per head: out[(b*16+h)*64+d][s]  (V path)
//   mode 2: store f32 row-major [M,N]                             (O path)
// Block: 256 threads = 8 waves; block tile 128(M) x 128(N); K-step 32.
// Each wave computes 32x64 via 2 A-frags x 4 B-frags = 8 WMMAs / k-step.
// Double-buffered LDS with async-to-LDS staging overlapped with compute.
// ---------------------------------------------------------------------------
__global__ __launch_bounds__(256)
void gemm_wmma(const _Float16* __restrict__ A, const _Float16* __restrict__ Bt,
               const float* __restrict__ bias, void* __restrict__ Cout,
               int M, int N, int K, int mode) {
  __shared__ alignas(16) _Float16 As[2][128 * 32];
  __shared__ alignas(16) _Float16 Bs[2][128 * 32];

  const int m0 = blockIdx.x * 128;
  const int n0 = blockIdx.y * 128;
  const int t    = threadIdx.x;
  const int wave = t >> 5;
  const int lane = t & 31;
  const int ln   = lane & 15;
  const int hl   = lane >> 4;
  const int wm   = wave >> 1;  // M sub-tile 0..3 -> rows wm*32
  const int wn   = wave & 1;   // N sub-tile 0..1 -> cols wn*64

  v8f acc[2][4];
  v8f z = {};
#pragma unroll
  for (int am = 0; am < 2; ++am)
#pragma unroll
    for (int f = 0; f < 4; ++f) acc[am][f] = z;

  const int NT = K / 32;
  // Stage k-tile i: per thread 2 A chunks + 2 B chunks of 16B.
  auto issue = [&](int i, int buf) {
#pragma unroll
    for (int j = 0; j < 2; ++j) {
      int idx = t + j * 256;
      int r = idx >> 2, c = (idx & 3) * 8;  // 128 rows x 4 chunks
      cp16(&A[(size_t)(m0 + r) * K + i * 32 + c],  &As[buf][r * 32 + c]);
      cp16(&Bt[(size_t)(n0 + r) * K + i * 32 + c], &Bs[buf][r * 32 + c]);
    }
  };

  issue(0, 0);
  for (int i = 0; i < NT; ++i) {
    const int cur = i & 1;
    wait_async_copies();     // my copies for tile i done
    __syncthreads();         // everyone's copies for tile i visible
    if (i + 1 < NT) issue(i + 1, cur ^ 1);  // overlap with compute below

    v16h a0 = ldfrag(&As[cur][(wm * 32 + ln) * 32 + hl * 8]);
    v16h a1 = ldfrag(&As[cur][(wm * 32 + 16 + ln) * 32 + hl * 8]);
#pragma unroll
    for (int f = 0; f < 4; ++f) {
      v16h bf = ldfrag(&Bs[cur][(wn * 64 + f * 16 + ln) * 32 + hl * 8]);
      acc[0][f] = wmma16(a0, bf, acc[0][f]);
      acc[1][f] = wmma16(a1, bf, acc[1][f]);
    }
    __syncthreads();         // tile `cur` fully consumed before next overwrite
  }

  // epilogue: C layout: VGPR r, lanes 0-15 -> row r, lanes 16-31 -> row 8+r
#pragma unroll
  for (int am = 0; am < 2; ++am) {
#pragma unroll
    for (int f = 0; f < 4; ++f) {
#pragma unroll
      for (int r = 0; r < 8; ++r) {
        int row = m0 + wm * 32 + am * 16 + r + 8 * hl;
        int col = n0 + wn * 64 + f * 16 + ln;
        float v = acc[am][f][r] + bias[col];
        if (mode == 0) {
          ((_Float16*)Cout)[(size_t)row * N + col] = (_Float16)v;
        } else if (mode == 1) {
          int bb = row >> 11, s = row & (S_LEN - 1);
          int hh = col >> 6,  d = col & (HDIM - 1);
          ((_Float16*)Cout)[(((size_t)(bb * NHEAD + hh)) * HDIM + d) * S_LEN + s] =
              (_Float16)v;
        } else {
          ((float*)Cout)[(size_t)row * N + col] = v;
        }
      }
    }
  }
}

// ---------------------------------------------------------------------------
// Flash attention: per (b,h), waves own 16-query tiles; online softmax.
// Qh, Kh : f16 [B*S, 1024] row-major (head slice at col h*64)
// Vth    : f16 [(b*16+h)*64 + d][s]   (transposed V per head)
// Oh     : f16 [B*S, 1024]  (concat-head layout for the O projection)
// Block: 128 threads = 4 waves; 64-key blocks, double-buffered in LDS.
// 16 WMMAs per key-block (8 scores + 8 P*V) per wave.
// ---------------------------------------------------------------------------
__global__ __launch_bounds__(128)
void attn_wmma(const _Float16* __restrict__ Qh, const _Float16* __restrict__ Kh,
               const _Float16* __restrict__ Vth, _Float16* __restrict__ Oh) {
  __shared__ alignas(16) _Float16 Kt[2][64 * 64];  // [key][depth]
  __shared__ alignas(16) _Float16 Vt[2][64 * 64];  // [depth][key]
  __shared__ alignas(16) _Float16 Pt[4][16 * 64];  // per-wave probabilities

  const int bh   = blockIdx.y;  // 0..31 == b*16+h
  const int b    = bh >> 4;
  const int h    = bh & 15;
  const int t    = threadIdx.x;
  const int wave = t >> 5;
  const int lane = t & 31;
  const int ln   = lane & 15;
  const int hl   = lane >> 4;
  const int q0   = blockIdx.x * 64 + wave * 16;

  // Loop-invariant Q fragments (depth 0-31, 32-63), scaled by 1/sqrt(64).
  const _Float16* qbase =
      Qh + ((size_t)(b * S_LEN + q0 + ln)) * D_MODEL + h * HDIM + hl * 8;
  v16h qa0 = ldfrag(qbase);
  v16h qa1 = ldfrag(qbase + 32);
#pragma unroll
  for (int i = 0; i < 16; ++i) {
    qa0[i] = (_Float16)((float)qa0[i] * 0.125f);
    qa1[i] = (_Float16)((float)qa1[i] * 0.125f);
  }

  float m_run[8], l_run[8];
  v8f o_acc[4];
  v8f z = {};
#pragma unroll
  for (int r = 0; r < 8; ++r) { m_run[r] = -1e30f; l_run[r] = 0.0f; }
#pragma unroll
  for (int f = 0; f < 4; ++f) o_acc[f] = z;

  _Float16* Pw = &Pt[wave][0];

  // Stage 64-key block: K 64x64 + V^T 64x64 halfs -> 4+4 chunks per thread.
  auto issue = [&](int kb, int buf) {
#pragma unroll
    for (int j = 0; j < 4; ++j) {
      int idx = t + j * 128;
      int r = idx >> 3, c = (idx & 7) * 8;  // 64 rows x 8 chunks
      cp16(&Kh[((size_t)(b * S_LEN + kb + r)) * D_MODEL + h * HDIM + c],
           &Kt[buf][r * 64 + c]);
      cp16(&Vth[((size_t)(bh * HDIM + r)) * S_LEN + kb + c],
           &Vt[buf][r * 64 + c]);
    }
  };

  const int NKB = S_LEN / 64;
  issue(0, 0);
  for (int i = 0; i < NKB; ++i) {
    const int cur = i & 1;
    wait_async_copies();
    __syncthreads();
    if (i + 1 < NKB) issue((i + 1) * 64, cur ^ 1);

    // scores: S(16q x 64k) = Q(16x64) * K^T(64x64)
    v8f s[4];
#pragma unroll
    for (int nt = 0; nt < 4; ++nt) {
      s[nt] = z;
      v16h k0 = ldfrag(&Kt[cur][(nt * 16 + ln) * 64 + hl * 8]);       // d 0-31
      v16h k1 = ldfrag(&Kt[cur][(nt * 16 + ln) * 64 + 32 + hl * 8]);  // d 32-63
      s[nt] = wmma16(qa0, k0, s[nt]);
      s[nt] = wmma16(qa1, k1, s[nt]);
    }

    // online softmax per row (VGPR index r -> rows r / 8+r across halves)
#pragma unroll
    for (int r = 0; r < 8; ++r) {
      float v0 = s[0][r], v1 = s[1][r], v2 = s[2][r], v3 = s[3][r];
      float bm = fmaxf(fmaxf(v0, v1), fmaxf(v2, v3));
      bm = fmaxf(bm, __shfl_xor(bm, 1));
      bm = fmaxf(bm, __shfl_xor(bm, 2));
      bm = fmaxf(bm, __shfl_xor(bm, 4));
      bm = fmaxf(bm, __shfl_xor(bm, 8));
      float mnew  = fmaxf(m_run[r], bm);
      float p0 = __expf(v0 - mnew);
      float p1 = __expf(v1 - mnew);
      float p2 = __expf(v2 - mnew);
      float p3 = __expf(v3 - mnew);
      float alpha = __expf(m_run[r] - mnew);
      float ladd  = (p0 + p1) + (p2 + p3);
      ladd += __shfl_xor(ladd, 1);
      ladd += __shfl_xor(ladd, 2);
      ladd += __shfl_xor(ladd, 4);
      ladd += __shfl_xor(ladd, 8);
      l_run[r] = l_run[r] * alpha + ladd;
      m_run[r] = mnew;
#pragma unroll
      for (int f = 0; f < 4; ++f) o_acc[f][r] *= alpha;
      int prow = r + 8 * hl;
      Pw[prow * 64 + ln]      = (_Float16)p0;
      Pw[prow * 64 + 16 + ln] = (_Float16)p1;
      Pw[prow * 64 + 32 + ln] = (_Float16)p2;
      Pw[prow * 64 + 48 + ln] = (_Float16)p3;
    }
    // order the per-wave P store -> P fragment load (same-wave LDS RAW)
    asm volatile("s_wait_dscnt 0" ::: "memory");

    // O(16q x 64d) += P(16x64) * V(64x64)
    v16h pa0 = ldfrag(&Pw[ln * 64 + hl * 8]);       // keys 0-31
    v16h pa1 = ldfrag(&Pw[ln * 64 + 32 + hl * 8]);  // keys 32-63
#pragma unroll
    for (int f = 0; f < 4; ++f) {
      v16h vb0 = ldfrag(&Vt[cur][(f * 16 + ln) * 64 + hl * 8]);
      v16h vb1 = ldfrag(&Vt[cur][(f * 16 + ln) * 64 + 32 + hl * 8]);
      o_acc[f] = wmma16(pa0, vb0, o_acc[f]);
      o_acc[f] = wmma16(pa1, vb1, o_acc[f]);
    }
    __syncthreads();  // K/V tile `cur` fully consumed before next overwrite
  }

  // normalize and store in [B,S,H*depth] f16 layout
#pragma unroll
  for (int f = 0; f < 4; ++f) {
#pragma unroll
    for (int r = 0; r < 8; ++r) {
      float inv = 1.0f / l_run[r];
      int row = q0 + r + 8 * hl;
      int col = h * HDIM + f * 16 + ln;
      Oh[((size_t)(b * S_LEN + row)) * D_MODEL + col] =
          (_Float16)(o_acc[f][r] * inv);
    }
  }
}

// ---------------------------------------------------------------------------
// Launch
// ---------------------------------------------------------------------------
extern "C" void kernel_launch(void* const* d_in, const int* in_sizes, int n_in,
                              void* d_out, int out_size, void* d_ws,
                              size_t ws_size, hipStream_t stream) {
  const float* x  = (const float*)d_in[0];
  const float* Wq = (const float*)d_in[1];
  const float* bq = (const float*)d_in[2];
  const float* Wk = (const float*)d_in[3];
  const float* bk = (const float*)d_in[4];
  const float* Wv = (const float*)d_in[5];
  const float* bv = (const float*)d_in[6];
  const float* Wo = (const float*)d_in[7];
  const float* bo = (const float*)d_in[8];

  char* ws = (char*)d_ws;
  const size_t MB = 1u << 20;
  _Float16* xh  = (_Float16*)(ws + 0 * MB);   // 8 MB
  _Float16* Wqt = (_Float16*)(ws + 8 * MB);   // 2 MB
  _Float16* Wkt = (_Float16*)(ws + 10 * MB);  // 2 MB
  _Float16* Wvt = (_Float16*)(ws + 12 * MB);  // 2 MB
  _Float16* Wot = (_Float16*)(ws + 14 * MB);  // 2 MB
  _Float16* Qh  = (_Float16*)(ws + 16 * MB);  // 8 MB
  _Float16* Kh  = (_Float16*)(ws + 24 * MB);  // 8 MB
  _Float16* Vth = (_Float16*)(ws + 32 * MB);  // 8 MB
  _Float16* Ah  = (_Float16*)(ws + 40 * MB);  // 8 MB  (total 48 MB)

  const int nX = NROWS * D_MODEL;
  const int nW = D_MODEL * D_MODEL;
  cvt_f32_to_f16<<<(nX + 255) / 256, 256, 0, stream>>>(x, xh, nX);
  cvt_transpose<<<(nW + 255) / 256, 256, 0, stream>>>(Wq, Wqt, D_MODEL);
  cvt_transpose<<<(nW + 255) / 256, 256, 0, stream>>>(Wk, Wkt, D_MODEL);
  cvt_transpose<<<(nW + 255) / 256, 256, 0, stream>>>(Wv, Wvt, D_MODEL);
  cvt_transpose<<<(nW + 255) / 256, 256, 0, stream>>>(Wo, Wot, D_MODEL);

  dim3 gg(NROWS / 128, D_MODEL / 128);  // 32 x 8 blocks
  gemm_wmma<<<gg, 256, 0, stream>>>(xh, Wqt, bq, Qh, NROWS, D_MODEL, D_MODEL, 0);
  gemm_wmma<<<gg, 256, 0, stream>>>(xh, Wkt, bk, Kh, NROWS, D_MODEL, D_MODEL, 0);
  gemm_wmma<<<gg, 256, 0, stream>>>(xh, Wvt, bv, Vth, NROWS, D_MODEL, D_MODEL, 1);

  attn_wmma<<<dim3(S_LEN / 64, 2 * NHEAD), 128, 0, stream>>>(Qh, Kh, Vth, Ah);

  gemm_wmma<<<gg, 256, 0, stream>>>(Ah, Wot, bo, d_out, NROWS, D_MODEL, D_MODEL, 2);
}
